// SpatialLinearAttention_91010357002897
// MI455X (gfx1250) — compile-verified, matrix operand order checked by reading, not run on hardware
//
#include <hip/hip_runtime.h>
#include <hip/hip_bf16.h>

// ---------------------------------------------------------------------------
// B=2, F=8, H=W=64, DIM=256, HEADS=4, DH=32
//   bf = 16 batch-frames, hw = 4096 tokens -> 4 heads x 1024 tokens
//   QKV  : (96x256)@(256x4096) per bf           -> wmma bf16, K-loop of 8
//   attn : S = Qs Ks^T / 32 ; P = softmax(S); O = P Vs   (per bf*head)
//   out  : (256x32)@(32x4096) per bf + bias     -> single wmma per tile
// All matmuls: v_wmma_f32_16x16x32_bf16 (dh = 32 == WMMA K).
// ---------------------------------------------------------------------------

typedef __attribute__((ext_vector_type(16))) __bf16 v16bf;
typedef __attribute__((ext_vector_type(8)))  __bf16 v8bf;
typedef __attribute__((ext_vector_type(8)))  float  v8f;

#define NBF   16
#define NHEAD 4
#define ATOK  1024
#define DH    32
#define HW    4096
#define CDIM  256

// K-map for 16-bit A/B fragments of v_wmma_*_16x16x32_bf16 (wave32):
// lanes 0-15 hold K = {0..7, 16..23}, lanes 16-31 hold K = {8..15, 24..31}
__device__ __forceinline__ int kmap(int i, int hi) {
  return (hi ? 8 : 0) + (i < 8 ? i : i + 8);
}

__device__ __forceinline__ v8f wmma_bf16(v16bf a, v16bf b, v8f c) {
  return __builtin_amdgcn_wmma_f32_16x16x32_bf16(false, a, false, b, (short)0, c,
                                                 false, false);
}

// Fragment from bf16 memory when the per-lane K run is contiguous:
// two 16-byte vector loads + shuffle (lowers to global_load_b128 x2).
__device__ __forceinline__ v16bf frag_bf16(const __bf16* row, int hi) {
  const __bf16* p = row + (hi ? 8 : 0);
  v8bf lo = *(const v8bf*)(p);
  v8bf hh = *(const v8bf*)(p + 16);
  return __builtin_shufflevector(lo, hh, 0, 1, 2, 3, 4, 5, 6, 7,
                                 8, 9, 10, 11, 12, 13, 14, 15);
}

// Same but converting from contiguous fp32 rows.
__device__ __forceinline__ v16bf frag_f32(const float* row, int hi) {
  const float* p = row + (hi ? 8 : 0);
  v16bf r;
#pragma unroll
  for (int i = 0; i < 8; ++i) r[i] = (__bf16)p[i];
#pragma unroll
  for (int i = 0; i < 8; ++i) r[8 + i] = (__bf16)p[16 + i];
  return r;
}

// ---------------------------------------------------------------------------
// Kernel 1: QKV projection. qkv[e,n] = sum_c w_qkv[e,c] * x[bf][c][n]
// grid = NBF * 6 * 256, one wave per 16x16 tile, 8 WMMA over 256 channels.
// Epilogue: each lane's 8 C elements are 8 consecutive dh -> one b128 store.
// ---------------------------------------------------------------------------
__global__ __launch_bounds__(32) void qkv_kernel(
    const float* __restrict__ x, const float* __restrict__ wqkv,
    __bf16* __restrict__ qws, __bf16* __restrict__ kws, __bf16* __restrict__ vws) {
  int bid   = blockIdx.x;
  int ntile = bid & 255;
  int tmp   = bid >> 8;
  int etile = tmp % 6;
  int bf    = tmp / 6;
  int lane  = threadIdx.x;
  int lm    = lane & 15;
  int hi    = lane >> 4;
  int e0    = etile * 16;
  int n0    = ntile * 16;
  int b     = bf >> 3, f = bf & 7;
  const float* xb = x + (size_t)b * (CDIM * 8 * HW) + (size_t)f * HW + n0 + lm;

  v8f acc = {};
  for (int k0 = 0; k0 < CDIM; k0 += 32) {
    // A: w_qkv rows are contiguous in k -> vectorized convert-load
    v16bf afrag = frag_f32(wqkv + (e0 + lm) * CDIM + k0, hi);
    // B: x is channel-major; element stride 8*HW floats, coalesced across lanes
    v16bf bfrag;
#pragma unroll
    for (int i = 0; i < 16; ++i)
      bfrag[i] = (__bf16)xb[(size_t)(k0 + kmap(i, hi)) * (8 * HW)];
    acc = wmma_bf16(afrag, bfrag, acc);
  }

  // C/D: lanes 0-15 -> M=r, N=lane ; lanes 16-31 -> M=8+r, N=lane-16
  int head = (n0 + lm) >> 10;
  int pos  = (n0 + lm) & (ATOK - 1);
  int erow = e0 + hi * 8;          // first row this lane writes
  int sel  = erow >> 5;            // 0=Q 1=K 2=V (8-row run never crosses)
  int dh0  = erow & 31;
  __bf16* dst = (sel == 0) ? qws : (sel == 1) ? kws : vws;
  v8bf pk;
#pragma unroll
  for (int r = 0; r < 8; ++r) pk[r] = (__bf16)acc[r];
  *(v8bf*)(dst + (((size_t)bf * NHEAD + head) * ATOK + pos) * DH + dh0) = pk;
}

// ---------------------------------------------------------------------------
// Kernel 2: attention per (bf*head, 16-row qtile). One wave per block.
// Pass 1: S(16x1024) via 64 WMMA -> LDS fp32 (scaled 1/32).
// Softmax: 2 lanes/row, float4 LDS traffic, shfl_xor pair reduction.
// Pass 2: O = P @ V via 64 WMMA; O transposed through LDS -> packed stores.
// grid = NBF*NHEAD*(ATOK/16) = 4096
// ---------------------------------------------------------------------------
__global__ __launch_bounds__(32) void attn_kernel(
    const __bf16* __restrict__ qws, const __bf16* __restrict__ kws,
    const __bf16* __restrict__ vws, __bf16* __restrict__ img) {
  __shared__ float sc[16 * ATOK];   // 64 KB score strip (reused for O transpose)

  int bid  = blockIdx.x;
  int qt   = bid & 63;
  int bfh  = bid >> 6;
  int lane = threadIdx.x;
  int lm   = lane & 15;
  int hi   = lane >> 4;
  int q0   = qt * 16;

  const __bf16* Qb = qws + (size_t)bfh * ATOK * DH;
  const __bf16* Kb = kws + (size_t)bfh * ATOK * DH;
  const __bf16* Vb = vws + (size_t)bfh * ATOK * DH;

  v16bf qa = frag_bf16(Qb + (q0 + lm) * DH, hi);   // A[m,d] = Qs[q0+m][d]

  // ---- Pass 1: scores -> LDS, scaled by 1/sqrt(1024) = 1/32 ----
  for (int kt = 0; kt < ATOK / 16; ++kt) {
    v16bf kb = frag_bf16(Kb + (kt * 16 + lm) * DH, hi);  // B[d,nk] = Ks[nk][d]
    v8f s = {};
    s = wmma_bf16(qa, kb, s);
#pragma unroll
    for (int r = 0; r < 8; ++r)
      sc[(r + hi * 8) * ATOK + kt * 16 + lm] = s[r] * 0.03125f;
  }
  __syncthreads();

  // ---- Softmax: row = lane/2, each of the pair handles 512 columns ----
  int row = lane >> 1;
  float4* rp = (float4*)(sc + row * ATOK + (lane & 1) * (ATOK / 2));
  float mx = -3.0e38f;
  for (int c = 0; c < ATOK / 8; ++c) {
    float4 t = rp[c];
    mx = fmaxf(mx, fmaxf(fmaxf(t.x, t.y), fmaxf(t.z, t.w)));
  }
  mx = fmaxf(mx, __shfl_xor(mx, 1, 32));
  float sum = 0.0f;
  for (int c = 0; c < ATOK / 8; ++c) {
    float4 t = rp[c];
    t.x = __expf(t.x - mx); t.y = __expf(t.y - mx);
    t.z = __expf(t.z - mx); t.w = __expf(t.w - mx);
    sum += t.x + t.y + t.z + t.w;
    rp[c] = t;
  }
  sum += __shfl_xor(sum, 1, 32);
  float myinv = 1.0f / sum;
  __syncthreads();
  float rlm = __shfl(myinv, lm << 1, 32);   // inv-sum of row lm (on lane 2*lm)

  // ---- Pass 2: O[m,e] = sum_n P[m,n] Vs[n][e], e in two 16-tiles ----
  v8f o0 = {}, o1 = {};
  for (int kb2 = 0; kb2 < ATOK / 32; ++kb2) {
    // P: contiguous fp32 runs in LDS, normalized + converted at frag build
    const float* prow = sc + lm * ATOK + kb2 * 32 + (hi ? 8 : 0);
    v16bf pa;
#pragma unroll
    for (int i = 0; i < 8; ++i) pa[i] = (__bf16)(prow[i] * rlm);
#pragma unroll
    for (int i = 0; i < 8; ++i) pa[8 + i] = (__bf16)(prow[16 + i] * rlm);
    // V: element stride DH, coalesced across lanes
    v16bf vb0, vb1;
#pragma unroll
    for (int i = 0; i < 16; ++i) {
      int kk = kb2 * 32 + kmap(i, hi);
      vb0[i] = Vb[kk * DH + lm];
      vb1[i] = Vb[kk * DH + 16 + lm];
    }
    o0 = wmma_bf16(pa, vb0, o0);
    o1 = wmma_bf16(pa, vb1, o1);
  }

  // ---- Transpose O through LDS, then packed 16B stores ----
  __syncthreads();
#pragma unroll
  for (int r = 0; r < 8; ++r) {
    int m = r + hi * 8;
    sc[m * DH + lm]      = o0[r];
    sc[m * DH + 16 + lm] = o1[r];
  }
  __syncthreads();
  int orow = lane >> 1, eb = (lane & 1) * 16;
  const float* op = sc + orow * DH + eb;
  v8bf p0, p1;
#pragma unroll
  for (int i = 0; i < 8; ++i) { p0[i] = (__bf16)op[i]; p1[i] = (__bf16)op[8 + i]; }
  __bf16* od = img + ((size_t)bfh * ATOK + q0 + orow) * DH + eb;
  *(v8bf*)(od)     = p0;
  *(v8bf*)(od + 8) = p1;
}

// ---------------------------------------------------------------------------
// Kernel 3: out[c,n] = sum_e w_out[c,e]*img[e,n] + bias[c]
// Single WMMA per 16x16 tile (K = 32). grid = NBF * 16 * 256.
// ---------------------------------------------------------------------------
__global__ __launch_bounds__(32) void out_kernel(
    const __bf16* __restrict__ img, const float* __restrict__ wout,
    const float* __restrict__ bias, float* __restrict__ out) {
  int bid  = blockIdx.x;
  int nt   = bid & 255;
  int tmp  = bid >> 8;
  int mt   = tmp & 15;
  int bf   = tmp >> 4;
  int lane = threadIdx.x;
  int lm   = lane & 15;
  int hi   = lane >> 4;
  int c0   = mt * 16;
  int n0   = nt * 16;

  v16bf afrag = frag_f32(wout + (c0 + lm) * DH, hi);                 // A[c,e]
  v16bf bfrag = frag_bf16(img + ((size_t)bf * HW + n0 + lm) * DH, hi); // B[e,n]
  v8f acc = {};
  acc = wmma_bf16(afrag, bfrag, acc);

  int b = bf >> 3, f = bf & 7;
#pragma unroll
  for (int r = 0; r < 8; ++r) {
    int c = c0 + r + hi * 8;
    out[(((size_t)b * CDIM + c) * 8 + f) * HW + n0 + lm] = acc[r] + bias[c];
  }
}

// ---------------------------------------------------------------------------
extern "C" void kernel_launch(void* const* d_in, const int* in_sizes, int n_in,
                              void* d_out, int out_size, void* d_ws, size_t ws_size,
                              hipStream_t stream) {
  const float* x    = (const float*)d_in[0];   // (2,256,8,64,64) fp32
  const float* wqkv = (const float*)d_in[1];   // (96,256)        fp32
  const float* wout = (const float*)d_in[2];   // (256,32)        fp32
  const float* bout = (const float*)d_in[3];   // (256,)          fp32
  float* out = (float*)d_out;

  char* ws = (char*)d_ws;
  const size_t SEG = (size_t)NBF * NHEAD * ATOK * DH * sizeof(__bf16);  // 4 MB
  __bf16* Q   = (__bf16*)(ws);
  __bf16* K   = (__bf16*)(ws + SEG);
  __bf16* V   = (__bf16*)(ws + 2 * SEG);
  __bf16* IMG = (__bf16*)(ws + 3 * SEG);

  qkv_kernel<<<NBF * 6 * 256, 32, 0, stream>>>(x, wqkv, Q, K, V);
  attn_kernel<<<NBF * NHEAD * (ATOK / 16), 32, 0, stream>>>(Q, K, V, IMG);
  out_kernel<<<NBF * 16 * 256, 32, 0, stream>>>(IMG, wout, bout, out);
}
